// SwinMSA_84353157693459
// MI455X (gfx1250) — compile-verified, matrix operand order checked by reading, not run on hardware
//
#include <hip/hip_runtime.h>
#include <hip/hip_bf16.h>
#include <math.h>

// Swin windowed MSA for MI455X (gfx1250, wave32).
// One wave = one window (S=10 tokens, E=96). Big GEMMs (95% of FLOPs) via
// v_wmma_f32_16x16x32_f16; tiny 10x10 attention + softmax in VALU.
// Memory-bound: ~504 MB HBM traffic -> ~22us floor at 23.3 TB/s.

#define WIN 10
#define HEADS 4
#define EDIM 96
#define DH 24
#define NWIN 4096
#define NBATCH 16
#define WPB 4   // waves (windows) per block

typedef __attribute__((ext_vector_type(16))) _Float16 v16h;
typedef __attribute__((ext_vector_type(8)))  _Float16 v8h;
typedef __attribute__((ext_vector_type(8)))  float    v8f;
typedef __attribute__((ext_vector_type(4)))  float    v4f;

// Load a 16-half WMMA fragment: halves [p, p+7] and [p+16, p+23].
// Matches CDNA5 16-bit A (16x32) and B (32x16) lane layouts when the caller
// passes p = row_base + kbase + (lane>=16 ? 8 : 0).
__device__ __forceinline__ v16h load_frag16(const _Float16* p) {
  v8h lo = *reinterpret_cast<const v8h*>(p);
  v8h hi = *reinterpret_cast<const v8h*>(p + 16);
  return __builtin_shufflevector(lo, hi, 0,1,2,3,4,5,6,7,8,9,10,11,12,13,14,15);
}

__device__ __forceinline__ v8f wmma_f16(v16h a, v16h b, v8f c) {
  return __builtin_amdgcn_wmma_f32_16x16x32_f16(false, a, false, b, (short)0, c,
                                                false, false);
}

// Prep: transpose + f32->f16 weights into [N][K] so B-fragments are two
// contiguous b128 loads. wt_qkv: [288][96], wt_o: [96][96].
__global__ void swin_prep_weights(const float* __restrict__ Wqkv,
                                  const float* __restrict__ Wo,
                                  _Float16* __restrict__ wt_qkv,
                                  _Float16* __restrict__ wt_o) {
  int i = blockIdx.x * blockDim.x + threadIdx.x;
  if (i < 288 * 96) {
    int n = i / 96, k = i % 96;
    wt_qkv[i] = (_Float16)Wqkv[k * 288 + n];
  }
  if (i < 96 * 96) {
    int n = i / 96, k = i % 96;
    wt_o[i] = (_Float16)Wo[k * 96 + n];
  }
}

__global__ __launch_bounds__(128) void swin_msa_kernel(
    const float* __restrict__ X,
    const float* __restrict__ b_qkv,
    const float* __restrict__ b_o,
    const float* __restrict__ pos_bias,
    const _Float16* __restrict__ wt_qkv,
    const _Float16* __restrict__ wt_o,
    float* __restrict__ Out) {
  // Per-wave LDS: X tile (reused as attention-output tile) and qkv tile
  // (reused as f32 output staging for the branch-free coalesced copy-out).
  __shared__ alignas(16) _Float16 lds_x[WPB][16 * EDIM];       // 16x96
  __shared__ alignas(16) _Float16 lds_qkv[WPB][16 * 3 * EDIM]; // 16x288

  const int wave = threadIdx.x >> 5;
  const int lane = threadIdx.x & 31;
  const int wid  = blockIdx.x * WPB + wave;   // global window id
  _Float16* xb   = lds_x[wave];
  _Float16* qkvb = lds_qkv[wave];

  // Zero pad rows 10..15 (stays zero: X load and attention write rows 0..9 only)
  for (int i = lane; i < 6 * EDIM; i += 32) xb[WIN * EDIM + i] = (_Float16)0.0f;

  // Load window X (10x96 f32 = 960 floats), convert to f16 rows 0..9.
  const float* xg = X + (size_t)wid * (WIN * EDIM);
  for (int i = lane; i < (WIN * EDIM) / 4; i += 32) {
    v4f v = reinterpret_cast<const v4f*>(xg)[i];
    xb[4 * i + 0] = (_Float16)v[0];
    xb[4 * i + 1] = (_Float16)v[1];
    xb[4 * i + 2] = (_Float16)v[2];
    xb[4 * i + 3] = (_Float16)v[3];
  }

  const int arow  = lane & 15;           // M row held by this lane
  const int koff  = (lane >> 4) * 8;     // K sub-offset per lane half
  const int rbase = (lane >> 4) << 3;    // C fragment: rows i+0 or i+8

  // ---------------- QKV projection: (16x96) x (96x288) ----------------
  v16h a0 = load_frag16(xb + arow * EDIM +  0 + koff);
  v16h a1 = load_frag16(xb + arow * EDIM + 32 + koff);
  v16h a2 = load_frag16(xb + arow * EDIM + 64 + koff);

  {
    const _Float16* wcol0 = wt_qkv + (lane & 15) * 96 + koff;  // tile 0 column
    v16h b0 = load_frag16(wcol0);
    v16h b1 = load_frag16(wcol0 + 32);
    v16h b2 = load_frag16(wcol0 + 64);
    for (int nt = 0; nt < 18; ++nt) {
      // Prefetch next tile's B fragments to overlap with this tile's WMMAs.
      const int nn = (nt + 1 < 18) ? (nt + 1) : 0;
      const _Float16* nwcol = wcol0 + nn * (16 * 96);
      v16h p0 = load_frag16(nwcol);
      v16h p1 = load_frag16(nwcol + 32);
      v16h p2 = load_frag16(nwcol + 64);

      v8f c = {};
      c = wmma_f16(a0, b0, c);
      c = wmma_f16(a1, b1, c);
      c = wmma_f16(a2, b2, c);

      const int col = nt * 16 + (lane & 15);
      const float bias = b_qkv[col];
      // Unconditional stores: rows 10..15 receive (0 + bias), never read.
#pragma unroll
      for (int i = 0; i < 8; ++i)
        qkvb[(i + rbase) * 288 + col] = (_Float16)(c[i] + bias);

      b0 = p0; b1 = p1; b2 = p2;
    }
  }

  // ---------------- Attention (per head 10x10, VALU + softmax) ----------------
  const bool last_win = ((wid & (NWIN - 1)) == (NWIN - 1));
  const float scale = 0.20412414523193154f;  // 1/sqrt(24)

  for (int idx = lane; idx < HEADS * WIN; idx += 32) {
    const int h  = idx / WIN;
    const int qi = idx % WIN;
    float qv[DH];
    const _Float16* qrow = qkvb + qi * 288 + h * DH;
#pragma unroll
    for (int d = 0; d < DH; ++d) qv[d] = (float)qrow[d];

    float s[WIN];
    float m = -1e30f;
    for (int kj = 0; kj < WIN; ++kj) {
      const _Float16* krow = qkvb + kj * 288 + EDIM + h * DH;
      float acc = 0.0f;
#pragma unroll
      for (int d = 0; d < DH; ++d) acc += qv[d] * (float)krow[d];
      acc = acc * scale + pos_bias[(kj - qi + WIN - 1) * HEADS + h];
      if (last_win && ((qi < 5) != (kj < 5))) acc -= 100.0f;
      s[kj] = acc;
      m = fmaxf(m, acc);
    }
    float sum = 0.0f;
    for (int kj = 0; kj < WIN; ++kj) { s[kj] = __expf(s[kj] - m); sum += s[kj]; }
    const float inv = 1.0f / sum;

    float o[DH];
#pragma unroll
    for (int d = 0; d < DH; ++d) o[d] = 0.0f;
    for (int kj = 0; kj < WIN; ++kj) {
      const _Float16* vrow = qkvb + kj * 288 + 2 * EDIM + h * DH;
      const float p = s[kj] * inv;
#pragma unroll
      for (int d = 0; d < DH; ++d) o[d] += p * (float)vrow[d];
    }
    _Float16* orow = xb + qi * EDIM + h * DH;   // reuse X tile as attn output
#pragma unroll
    for (int d = 0; d < DH; ++d) orow[d] = (_Float16)o[d];
  }

  // ---------------- Output projection: (16x96) x (96x96) ----------------
  v16h oa0 = load_frag16(xb + arow * EDIM +  0 + koff);
  v16h oa1 = load_frag16(xb + arow * EDIM + 32 + koff);
  v16h oa2 = load_frag16(xb + arow * EDIM + 64 + koff);

  // Stage 16x96 f32 results in LDS (reusing the now-dead qkv tile), then do a
  // branch-free, fully coalesced copy-out of rows 0..9.
  float* ob = reinterpret_cast<float*>(qkvb);

  {
    const _Float16* wcol0 = wt_o + (lane & 15) * 96 + koff;
    v16h b0 = load_frag16(wcol0);
    v16h b1 = load_frag16(wcol0 + 32);
    v16h b2 = load_frag16(wcol0 + 64);
    for (int nt = 0; nt < 6; ++nt) {
      const int nn = (nt + 1 < 6) ? (nt + 1) : 0;
      const _Float16* nwcol = wcol0 + nn * (16 * 96);
      v16h p0 = load_frag16(nwcol);
      v16h p1 = load_frag16(nwcol + 32);
      v16h p2 = load_frag16(nwcol + 64);

      v8f c = {};
      c = wmma_f16(oa0, b0, c);
      c = wmma_f16(oa1, b1, c);
      c = wmma_f16(oa2, b2, c);

      const int col = nt * 16 + (lane & 15);
      const float bias = b_o[col];
#pragma unroll
      for (int i = 0; i < 8; ++i)
        ob[(i + rbase) * EDIM + col] = c[i] + bias;

      b0 = p0; b1 = p1; b2 = p2;
    }
  }

  // Coalesced copy-out: 240 float4s = rows 0..9 of the staged tile.
  float* og = Out + (size_t)wid * (WIN * EDIM);
  for (int i = lane; i < (WIN * EDIM) / 4; i += 32) {
    reinterpret_cast<v4f*>(og)[i] = reinterpret_cast<const v4f*>(ob)[i];
  }
}

extern "C" void kernel_launch(void* const* d_in, const int* in_sizes, int n_in,
                              void* d_out, int out_size, void* d_ws, size_t ws_size,
                              hipStream_t stream) {
  const float* X        = (const float*)d_in[0];
  const float* W_qkv    = (const float*)d_in[1];
  const float* b_qkv    = (const float*)d_in[2];
  const float* W_o      = (const float*)d_in[3];
  const float* b_o      = (const float*)d_in[4];
  const float* pos_bias = (const float*)d_in[5];

  _Float16* wt_qkv = (_Float16*)d_ws;           // 288*96 f16
  _Float16* wt_o   = wt_qkv + 288 * 96;         // 96*96 f16

  swin_prep_weights<<<(288 * 96 + 255) / 256, 256, 0, stream>>>(
      W_qkv, W_o, wt_qkv, wt_o);

  const int nwin = NBATCH * NWIN;               // 65536 windows
  swin_msa_kernel<<<nwin / WPB, WPB * 32, 0, stream>>>(
      X, b_qkv, b_o, pos_bias, wt_qkv, wt_o, (float*)d_out);
}
